// MultiHeadedAttention_17360257810683
// MI455X (gfx1250) — compile-verified, matrix operand order checked by reading, not run on hardware
//
#include <hip/hip_runtime.h>
#include <hip/hip_bf16.h>

typedef unsigned short ushort_t;
typedef __attribute__((ext_vector_type(16))) __bf16 v16bf;
typedef __attribute__((ext_vector_type(8)))  float  v8f;

#define B_  16
#define S_  400
#define D_  1024
#define H_  16
#define DH_ 64
#define M_  (B_ * S_)   // 6400

#define BM 128
#define BN 64
#define BK 32

// ---- CDNA5 async global->LDS path (gate on toolchain support, fall back to sync)
#if defined(__has_builtin)
#  if __has_builtin(__builtin_amdgcn_global_load_async_to_lds_b128) && \
      __has_builtin(__builtin_amdgcn_s_wait_asynccnt)
#    define USE_ASYNC_LDS 1
#  endif
#endif
#ifndef USE_ASYNC_LDS
#  define USE_ASYNC_LDS 0
#endif

// builtin prototype (from hipcc sema): (int4 AS1*, int4 AS3*, imm offset, imm cpol)
typedef int vint4_t __attribute__((vector_size(16)));
#define AS_GLOBAL(p) ((__attribute__((address_space(1))) vint4_t*)(p))
#define AS_LDS(p)    ((__attribute__((address_space(3))) vint4_t*)(p))

__device__ __forceinline__ void wait_async_all() {
#if USE_ASYNC_LDS
    __builtin_amdgcn_s_wait_asynccnt(0);
#endif
}

union Frag { uint4 u[2]; v16bf v; };

__device__ __forceinline__ ushort_t f2bf(float f) {
    unsigned int u = __float_as_uint(f);
    u += 0x7FFFu + ((u >> 16) & 1u);
    return (ushort_t)(u >> 16);
}

// ---------------------------------------------------------------- convert
__global__ __launch_bounds__(256) void f32_to_bf16_kernel(const float* __restrict__ s,
                                                          ushort_t* __restrict__ d, long n) {
    long i = (long)blockIdx.x * blockDim.x + threadIdx.x;
    long stride = (long)gridDim.x * blockDim.x;
    for (; i < n; i += stride) d[i] = f2bf(s[i]);
}

// ---------------------------------------------------------------- layernorm (+optional PE)
__global__ __launch_bounds__(256) void ln_kernel(const float* __restrict__ x,
                                                 const float* __restrict__ g,
                                                 const float* __restrict__ bt,
                                                 float* __restrict__ outF,
                                                 ushort_t* __restrict__ outB,
                                                 int addPE) {
    __shared__ float sred[256];
    __shared__ float s_mu, s_rstd;
    const int row = blockIdx.x;
    const int tid = threadIdx.x;
    const long base = (long)row * D_;
    float lv[4];
    float ls = 0.f;
    #pragma unroll
    for (int j = 0; j < 4; ++j) { lv[j] = x[base + tid + 256 * j]; ls += lv[j]; }
    sred[tid] = ls; __syncthreads();
    for (int off = 128; off > 0; off >>= 1) {
        if (tid < off) sred[tid] += sred[tid + off];
        __syncthreads();
    }
    if (tid == 0) s_mu = sred[0] * (1.0f / D_);
    __syncthreads();
    const float mu = s_mu;
    float lq = 0.f;
    #pragma unroll
    for (int j = 0; j < 4; ++j) { float d = lv[j] - mu; lq += d * d; }
    sred[tid] = lq; __syncthreads();
    for (int off = 128; off > 0; off >>= 1) {
        if (tid < off) sred[tid] += sred[tid + off];
        __syncthreads();
    }
    if (tid == 0) s_rstd = rsqrtf(sred[0] * (1.0f / D_) + 1e-5f);
    __syncthreads();
    const float rstd = s_rstd;
    const int srow = row % S_;  // position for PE
    #pragma unroll
    for (int j = 0; j < 4; ++j) {
        int c = tid + 256 * j;
        float v = g[c] * (lv[j] - mu) * rstd + bt[c];
        if (addPE) {
            int i2 = c >> 1;
            float ang = (float)srow * __expf(-(2.0f * (float)i2 / (float)D_) * 9.210340372f);
            v += (c & 1) ? __cosf(ang) : __sinf(ang);
        }
        if (outF) outF[base + c] = v;
        if (outB) outB[base + c] = f2bf(v);
    }
}

// ---------------------------------------------------------------- generic bf16 WMMA GEMM
struct GemmP {
    const ushort_t* A;  long lda;
    const ushort_t* Bm; long ldb;
    float* Cf; ushort_t* Cb; long ldc;
    const float* bias;
    const float* Res; long ldr;
    int M, N, K;
    float alpha;
    int relu, transB;
    int zdiv;
    long sA1, sA2, sB1, sB2, sC1, sC2;
};

__global__ __launch_bounds__(256) void gemm_wmma_kernel(GemmP p) {
    // double-buffered tiles for async global->LDS pipelining
    __shared__ __align__(16) ushort_t As[2][BM * BK];   // [m][k]
    __shared__ __align__(16) ushort_t Bs[2][BN * BK];   // [n][k]  (B^T staged)

    const int tid = threadIdx.x;
    const int z  = blockIdx.z;
    const int zb = z / p.zdiv, zh = z % p.zdiv;
    const long aB = (long)zb * p.sA1 + (long)zh * p.sA2;
    const long bB = (long)zb * p.sB1 + (long)zh * p.sB2;
    const long cB = (long)zb * p.sC1 + (long)zh * p.sC2;

    const int rowBase = blockIdx.y * BM;
    const int colBase = blockIdx.x * BN;
    const int wave = tid >> 5, lane = tid & 31;
    const int wm = wave & 3, wn = wave >> 2;       // 4 (M) x 2 (N) waves
    const int half = lane >> 4, lrow = lane & 15;

    // stage one BK-slice (k0) into buffer `buf`
    auto stage = [&](int buf, int k0) {
        // ---- A tile: 128x32 bf16, 512 chunks of 8, 2 per thread
        #pragma unroll
        for (int i = 0; i < 2; ++i) {
            int c = tid + i * 256;
            int r = c >> 2, kc = (c & 3) * 8;
            int grow = rowBase + r;
            ushort_t* dst = &As[buf][r * BK + kc];
            const ushort_t* src = p.A + aB + (long)grow * p.lda + k0 + kc;
            if (grow < p.M && (k0 + kc + 8) <= p.K) {
#if USE_ASYNC_LDS
                __builtin_amdgcn_global_load_async_to_lds_b128(
                    AS_GLOBAL(src), AS_LDS(dst), 0, 0);
#else
                *(uint4*)dst = *(const uint4*)src;
#endif
            } else {
                #pragma unroll
                for (int e = 0; e < 8; ++e) {
                    int k = k0 + kc + e;
                    dst[e] = (grow < p.M && k < p.K)
                                 ? p.A[aB + (long)grow * p.lda + k] : (ushort_t)0;
                }
            }
        }
        // ---- B tile transposed: Bs[n][k], 256 chunks of 8, 1 per thread
        {
            int n = tid >> 2, kc = (tid & 3) * 8;
            int gcol = colBase + n;
            ushort_t* dst = &Bs[buf][n * BK + kc];
            if (p.transB) {  // B[k][n] = Bm[n*ldb + k]  -> contiguous in k
                const ushort_t* src = p.Bm + bB + (long)gcol * p.ldb + k0 + kc;
                if (gcol < p.N && (k0 + kc + 8) <= p.K) {
#if USE_ASYNC_LDS
                    __builtin_amdgcn_global_load_async_to_lds_b128(
                        AS_GLOBAL(src), AS_LDS(dst), 0, 0);
#else
                    *(uint4*)dst = *(const uint4*)src;
#endif
                } else {
                    #pragma unroll
                    for (int e = 0; e < 8; ++e) {
                        int k = k0 + kc + e;
                        dst[e] = (gcol < p.N && k < p.K)
                                     ? p.Bm[bB + (long)gcol * p.ldb + k] : (ushort_t)0;
                    }
                }
            } else {         // B row-major [k][n] -> strided gather (sync path)
                #pragma unroll
                for (int e = 0; e < 8; ++e) {
                    int k = k0 + kc + e;
                    dst[e] = (gcol < p.N && k < p.K)
                                 ? p.Bm[bB + (long)k * p.ldb + gcol] : (ushort_t)0;
                }
            }
        }
    };

    const v8f vzero = {0.f, 0.f, 0.f, 0.f, 0.f, 0.f, 0.f, 0.f};
    v8f acc[2][2];
    acc[0][0] = vzero; acc[0][1] = vzero; acc[1][0] = vzero; acc[1][1] = vzero;

    const int nk = (p.K + BK - 1) / BK;

    // prologue: fill buffer 0
    stage(0, 0);
    wait_async_all();
    __syncthreads();

    for (int i = 0; i < nk; ++i) {
        const int cur = i & 1;
        const ushort_t* Ac = As[cur];
        const ushort_t* Bc = Bs[cur];

        // prefetch next slice into the other buffer while we compute
        if (i + 1 < nk) stage(cur ^ 1, (i + 1) * BK);

        // ---- fragments per ISA 16-bit layouts:
        // lanes 0-15 hold K{0..7,16..23}, lanes 16-31 hold K{8..15,24..31}
        Frag aF[2], bF[2];
        #pragma unroll
        for (int m16 = 0; m16 < 2; ++m16) {
            const ushort_t* pA = &Ac[(wm * 32 + m16 * 16 + lrow) * BK + half * 8];
            aF[m16].u[0] = *(const uint4*)pA;
            aF[m16].u[1] = *(const uint4*)(pA + 16);
        }
        #pragma unroll
        for (int n16 = 0; n16 < 2; ++n16) {
            const ushort_t* pB = &Bc[(wn * 32 + n16 * 16 + lrow) * BK + half * 8];
            bF[n16].u[0] = *(const uint4*)pB;
            bF[n16].u[1] = *(const uint4*)(pB + 16);
        }
        #pragma unroll
        for (int m16 = 0; m16 < 2; ++m16)
            #pragma unroll
            for (int n16 = 0; n16 < 2; ++n16)
                acc[m16][n16] = __builtin_amdgcn_wmma_f32_16x16x32_bf16(
                    false, aF[m16].v, false, bF[n16].v,
                    (short)0, acc[m16][n16], false, false);

        if (i + 1 < nk) {
            wait_async_all();   // staged slice fully in LDS
            __syncthreads();    // ds-store fallback + all frag reads ordered
        }
    }

    // ---- epilogue: C VGPR r -> row r (lanes 0-15) / row r+8 (lanes 16-31)
    #pragma unroll
    for (int m16 = 0; m16 < 2; ++m16) {
        #pragma unroll
        for (int n16 = 0; n16 < 2; ++n16) {
            #pragma unroll
            for (int r = 0; r < 8; ++r) {
                int row = rowBase + wm * 32 + m16 * 16 + half * 8 + r;
                int col = colBase + wn * 32 + n16 * 16 + lrow;
                if (row < p.M && col < p.N) {
                    float v = p.alpha * acc[m16][n16][r];
                    if (p.bias) v += p.bias[col];
                    if (p.Res)  v += p.Res[(long)row * p.ldr + col];
                    if (p.relu) v = fmaxf(v, 0.f);
                    long ci = cB + (long)row * p.ldc + col;
                    if (p.Cf) p.Cf[ci] = v;
                    if (p.Cb) p.Cb[ci] = f2bf(v);
                }
            }
        }
    }
}

// ---------------------------------------------------------------- row softmax -> bf16 probs
// grid = (S_, B_*H_), block 256.  scores: [z][S][S] f32, mask: [b][S][S] bool
__global__ __launch_bounds__(256) void softmax_kernel(const float* __restrict__ scores,
                                                      const unsigned char* __restrict__ mask,
                                                      ushort_t* __restrict__ attn) {
    __shared__ float sred[256];
    const int q = blockIdx.x;
    const int z = blockIdx.y;
    const int b = z >> 4;
    const int tid = threadIdx.x;
    const long rowOff = ((long)z * S_ + q) * S_;
    const long mOff   = ((long)b * S_ + q) * S_;

    const int i0 = tid;            // always < 400
    const int i1 = tid + 256;
    const bool in1 = (i1 < S_);

    float v0 = scores[rowOff + i0];
    if (mask[mOff + i0]) v0 = -1e18f;
    float v1 = -3.4e38f;
    if (in1) { v1 = scores[rowOff + i1]; if (mask[mOff + i1]) v1 = -1e18f; }

    sred[tid] = fmaxf(v0, v1); __syncthreads();
    for (int off = 128; off > 0; off >>= 1) {
        if (tid < off) sred[tid] = fmaxf(sred[tid], sred[tid + off]);
        __syncthreads();
    }
    const float mx = sred[0]; __syncthreads();

    float e0 = __expf(v0 - mx);
    float e1 = in1 ? __expf(v1 - mx) : 0.f;
    sred[tid] = e0 + e1; __syncthreads();
    for (int off = 128; off > 0; off >>= 1) {
        if (tid < off) sred[tid] += sred[tid + off];
        __syncthreads();
    }
    const float inv = 1.0f / sred[0];

    attn[rowOff + i0] = f2bf(e0 * inv);
    if (in1) attn[rowOff + i1] = f2bf(e1 * inv);
}

// ---------------------------------------------------------------- host-side helpers
static void launch_gemm(hipStream_t stream,
                        const ushort_t* A, long lda,
                        const ushort_t* Bm, long ldb, int transB,
                        float* Cf, ushort_t* Cb, long ldc,
                        const float* bias, const float* Res, long ldr,
                        int M, int N, int K, float alpha, int relu,
                        int batches, int zdiv,
                        long sA1, long sA2, long sB1, long sB2, long sC1, long sC2) {
    GemmP p;
    p.A = A; p.lda = lda; p.Bm = Bm; p.ldb = ldb;
    p.Cf = Cf; p.Cb = Cb; p.ldc = ldc;
    p.bias = bias; p.Res = Res; p.ldr = ldr;
    p.M = M; p.N = N; p.K = K; p.alpha = alpha; p.relu = relu; p.transB = transB;
    p.zdiv = zdiv;
    p.sA1 = sA1; p.sA2 = sA2; p.sB1 = sB1; p.sB2 = sB2; p.sC1 = sC1; p.sC2 = sC2;
    dim3 grid((N + BN - 1) / BN, (M + BM - 1) / BM, batches);
    gemm_wmma_kernel<<<grid, 256, 0, stream>>>(p);
}

extern "C" void kernel_launch(void* const* d_in, const int* in_sizes, int n_in,
                              void* d_out, int out_size, void* d_ws, size_t ws_size,
                              hipStream_t stream) {
    (void)in_sizes; (void)n_in; (void)out_size; (void)ws_size;

    const float*         inp    = (const float*)d_in[0];
    const unsigned char* amask  = (const unsigned char*)d_in[1];
    const float*         w_qkv  = (const float*)d_in[2];
    const float*         w_fin  = (const float*)d_in[3];
    const float*         w1     = (const float*)d_in[4];
    const float*         b1     = (const float*)d_in[5];
    const float*         w2     = (const float*)d_in[6];
    const float*         b2     = (const float*)d_in[7];
    const float*         g_mh   = (const float*)d_in[8];
    const float*         bt_mh  = (const float*)d_in[9];
    const float*         g_ffn  = (const float*)d_in[10];
    const float*         bt_ffn = (const float*)d_in[11];
    const float*         g_out  = (const float*)d_in[12];
    const float*         bt_out = (const float*)d_in[13];
    float* out = (float*)d_out;

    // ---- workspace carve-up (256B aligned)
    char* w = (char*)d_ws;
    size_t o = 0;
    auto alloc = [&](size_t bytes) -> size_t {
        size_t r = o; o = (o + bytes + 255) & ~(size_t)255; return r;
    };
    float*    xf     = (float*)   (w + alloc((size_t)M_ * D_ * 4));       // LN1+PE residual
    ushort_t* xb     = (ushort_t*)(w + alloc((size_t)M_ * D_ * 2));
    ushort_t* wqkvb  = (ushort_t*)(w + alloc((size_t)D_ * 3 * D_ * 2));
    ushort_t* wfb    = (ushort_t*)(w + alloc((size_t)D_ * D_ * 2));
    ushort_t* w1b    = (ushort_t*)(w + alloc((size_t)D_ * 4 * D_ * 2));
    ushort_t* w2b    = (ushort_t*)(w + alloc((size_t)4 * D_ * D_ * 2));
    ushort_t* qkvb   = (ushort_t*)(w + alloc((size_t)M_ * 3 * D_ * 2));
    float*    scores = (float*)   (w + alloc((size_t)B_ * H_ * S_ * S_ * 4));
    ushort_t* attnb  = (ushort_t*)(w + alloc((size_t)B_ * H_ * S_ * S_ * 2));
    ushort_t* ctxb   = (ushort_t*)(w + alloc((size_t)M_ * D_ * 2));
    float*    aoutf  = (float*)   (w + alloc((size_t)M_ * D_ * 4));       // attn out + residual
    float*    yf     = (float*)   (w + alloc((size_t)M_ * D_ * 4));       // LN2 out (residual_ffn)
    ushort_t* yb     = (ushort_t*)(w + alloc((size_t)M_ * D_ * 2));
    ushort_t* hb     = (ushort_t*)(w + alloc((size_t)M_ * 4 * D_ * 2));   // relu(ffn1)
    float*    zf     = (float*)   (w + alloc((size_t)M_ * D_ * 4));       // pre-final-LN

    // 1) LN(inp) + PE -> xf (f32 residual), xb (bf16)
    ln_kernel<<<M_, 256, 0, stream>>>(inp, g_mh, bt_mh, xf, xb, 1);

    // 2) weights -> bf16
    f32_to_bf16_kernel<<<2048, 256, 0, stream>>>(w_qkv, wqkvb, (long)D_ * 3 * D_);
    f32_to_bf16_kernel<<<2048, 256, 0, stream>>>(w_fin, wfb,   (long)D_ * D_);
    f32_to_bf16_kernel<<<2048, 256, 0, stream>>>(w1,    w1b,   (long)D_ * 4 * D_);
    f32_to_bf16_kernel<<<2048, 256, 0, stream>>>(w2,    w2b,   (long)4 * D_ * D_);

    // 3) QKV: [6400,1024] x [1024,3072] -> bf16 [6400,3072]
    launch_gemm(stream, xb, D_, wqkvb, 3 * D_, 0,
                nullptr, qkvb, 3 * D_, nullptr, nullptr, 0,
                M_, 3 * D_, D_, 1.0f, 0,
                1, 1, 0, 0, 0, 0, 0, 0);

    // 4) scores[b,h] = (q/8) @ k^T : batched 256, A=q slice, B=k slice (transB)
    const long qkvRowB = (long)S_ * 3 * D_;   // per-batch-b row block in qkv
    launch_gemm(stream, qkvb /*q*/, 3 * D_,
                qkvb + D_ /*k*/, 3 * D_, 1,
                scores, nullptr, S_, nullptr, nullptr, 0,
                S_, S_, DH_, 0.125f, 0,
                B_ * H_, H_,
                qkvRowB, DH_,                 // A: +b*S*3D, +h*64
                qkvRowB, DH_,                 // B: same pattern (k base already +1024)
                (long)H_ * S_ * S_, (long)S_ * S_);  // C: z = b*16+h -> [z][S][S]

    // 5) softmax rows -> bf16 probs
    {
        dim3 g(S_, B_ * H_);
        softmax_kernel<<<g, 256, 0, stream>>>(scores, amask, attnb);
    }

    // 6) ctx[b,h] = attn @ v : batched 256 -> ctxb[6400,1024] at col h*64
    launch_gemm(stream, attnb, S_,
                qkvb + 2 * D_ /*v*/, 3 * D_, 0,
                nullptr, ctxb, D_, nullptr, nullptr, 0,
                S_, DH_, S_, 1.0f, 0,
                B_ * H_, H_,
                (long)H_ * S_ * S_, (long)S_ * S_,  // A: attn [z][S][S]
                qkvRowB, DH_,                       // B: v slice
                (long)S_ * D_, DH_);                // C: +b*S*D, +h*64

    // 7) out = ctx @ w_final + residual(xf) -> aoutf (f32)
    launch_gemm(stream, ctxb, D_, wfb, D_, 0,
                aoutf, nullptr, D_, nullptr, xf, D_,
                M_, D_, D_, 1.0f, 0,
                1, 1, 0, 0, 0, 0, 0, 0);

    // 8) y = LN(aoutf) -> yf (residual_ffn, f32), yb (bf16)
    ln_kernel<<<M_, 256, 0, stream>>>(aoutf, g_ffn, bt_ffn, yf, yb, 0);

    // 9) h = relu(y @ w1 + b1) -> hb (bf16 [6400,4096])
    launch_gemm(stream, yb, D_, w1b, 4 * D_, 0,
                nullptr, hb, 4 * D_, b1, nullptr, 0,
                M_, 4 * D_, D_, 1.0f, 1,
                1, 1, 0, 0, 0, 0, 0, 0);

    // 10) z = h @ w2 + b2 + residual(yf) -> zf (f32)
    launch_gemm(stream, hb, 4 * D_, w2b, D_, 0,
                zf, nullptr, D_, b2, yf, D_,
                M_, D_, 4 * D_, 1.0f, 0,
                1, 1, 0, 0, 0, 0, 0, 0);

    // 11) final LN -> d_out (f32)
    ln_kernel<<<M_, 256, 0, stream>>>(zf, g_out, bt_out, out, nullptr, 0);
}
